// LinearLLM_63376537420349
// MI455X (gfx1250) — matrix operand order repaired and checked by reading, not compiled, not run on hardware
//
#include <hip/hip_runtime.h>

// ---- types for WMMA / staging ----
typedef __bf16  v16bf __attribute__((ext_vector_type(16)));
typedef __bf16  bf16x8 __attribute__((ext_vector_type(8)));
typedef float   v8f   __attribute__((ext_vector_type(8)));
typedef float   f32x8 __attribute__((ext_vector_type(8)));
typedef float   f32x4 __attribute__((ext_vector_type(4)));

#define L1    513
#define EMB   64
#define NTOK  6
#define BATCH 128
#define NCOL  (L1 * NTOK)   // 3078 flattened (t,v) columns
#define KDIM  (L1 * EMB)    // 32832 flattened (s,w) reduction dim
#define NT    32            // N-tile per workgroup (2 WMMA column tiles)
#define KT    64            // K-chunk staged in LDS (exactly one s per chunk)
#define KP    72            // padded LDS row stride (elements) to spread banks

// Kernel 1: X[b][s*64+w] = bf16(embedding[src[b,s]][w])   (8.4 MB in d_ws)
__global__ void gather_x_kernel(const int* __restrict__ src,
                                const float* __restrict__ emb,
                                __bf16* __restrict__ X) {
    int tid = blockIdx.x * blockDim.x + threadIdx.x;   // over B*L1 = 65664
    if (tid >= BATCH * L1) return;
    int token = src[tid];
    const float* e = emb + (size_t)token * EMB;        // 6x64 table: cache-hot
    __bf16* xr = X + (size_t)tid * EMB;                // (b*513+s)*64 == b*32832+s*64
#pragma unroll
    for (int g = 0; g < EMB; g += 8) {
        f32x8 w;
#pragma unroll
        for (int j = 0; j < 8; ++j) w[j] = e[g + j];
        bf16x8 t = __builtin_convertvector(w, bf16x8); // 4x v_cvt_pk_bf16_f32
        *(bf16x8*)(xr + g) = t;                        // 16B aligned store
    }
}

// Kernel 2: masked GEMM via v_wmma_f32_16x16x32_bf16, double-buffered LDS.
// Block = 256 threads = 8 waves; wave m owns batch rows [16m,16m+16).
// Block owns 32 output columns n0..n0+31 (n = t*6+v); weight read once chip-wide.
__global__ void __launch_bounds__(256)
gemm_wmma_kernel(const __bf16* __restrict__ X,
                 const float* __restrict__ W,
                 const float* __restrict__ bias,
                 float* __restrict__ out) {
    __shared__ __align__(16) __bf16 WT[2][NT * KP];    // 2 x 4.6 KB bf16 tiles

    const int n0   = blockIdx.x * NT;
    const int tid  = threadIdx.x;
    const int lane = tid & 31;
    const int wave = tid >> 5;      // 0..7 -> M tile
    const int h    = lane >> 4;     // lane half (K-half selector)
    const int lm   = lane & 15;     // row/col within fragment

    // causal K bound for this column tile: only s <= t_max rows are ever touched
    int t_hi = (n0 + NT - 1) / NTOK;
    if (t_hi > L1 - 1) t_hi = L1 - 1;
    const int nchunk = t_hi + 1;    // number of KT=64 chunks (one s per chunk)

    // staging assignment: lane -> column, wave -> 8 consecutive k rows.
    // Each load instruction is 32 lanes x contiguous columns = coalesced 128B.
    // Each thread's 8 staged elements are contiguous in WT -> one ds_store_b128.
    const int sn   = tid & 31;       // column within tile
    const int kg   = (tid >> 5) * 8; // base k row for this thread
    const int gn   = n0 + sn;        // global column
    const bool ncol_ok = (gn < NCOL);
    const int tcol = ncol_ok ? (gn / NTOK) : -1; // mask: keep iff s <= tcol

    // Running base pointer: loads become base + immediate offsets (i*NCOL*4 B),
    // advanced once per chunk -- no per-element 64-bit address MADs.
    const size_t CHUNK_STEP = (size_t)KT * NCOL;
    const float* wp = W + (size_t)kg * NCOL + gn;    // chunk 0 base

    __bf16* const ldst = &WT[0][sn * KP + kg];       // this thread's LDS slot
    const ptrdiff_t BUFSTEP = NT * KP;               // buffer stride (elements)

    // Fully branchless masked stage of chunk s into buffer `buf`:
    // unconditional clause of 8 loads from a safe base, cndmask the values.
    auto stage = [&](const float* wpc, int s, int buf) {
        const bool keep = (s <= tcol);               // causal mask, per column
        const float* wsafe = keep ? wpc : W;         // 64-bit cndmask, no branch
        f32x8 wv;
#pragma unroll
        for (int i = 0; i < 8; ++i) {
            wv[i] = wsafe[(size_t)i * NCOL];         // base + imm: clause of b32
        }
        f32x8 z = {};
        wv = keep ? wv : z;                          // value select (cndmask)
        bf16x8 pk = __builtin_convertvector(wv, bf16x8); // 4x v_cvt_pk_bf16_f32
        *(bf16x8*)(ldst + buf * BUFSTEP) = pk;       // single 16B LDS store
    };

    v8f acc0 = {};
    v8f acc1 = {};

    stage(wp, 0, 0);
    __syncthreads();

    int buf = 0;
    for (int s = 0; s < nchunk; ++s) {
        const int k0 = s * KT;
        // ---- stage next chunk into the other buffer while we compute ----
        if (s + 1 < nchunk) {
            stage(wp + CHUNK_STEP, s + 1, buf ^ 1);
            // speculative prefetch two chunks ahead (safe, branchless address)
            const float* pf = (s + 2 <= tcol) ? (wp + 2 * CHUNK_STEP) : W;
            __builtin_prefetch(pf, 0, 3);
        }
        wp += CHUNK_STEP;

        // ---- consume current buffer: 2 WMMA K-steps x 2 N-subtiles ----
#pragma unroll
        for (int ks = 0; ks < KT; ks += 32) {
            // A fragment (16x32 bf16): lane lm = batch row, two 8-elem K runs
            const __bf16* xrow =
                X + (size_t)(wave * 16 + lm) * KDIM + (k0 + ks);
            union { v16bf v; f32x4 q[2]; } A;
            A.q[0] = *(const f32x4*)(xrow + h * 8);        // K = ks + h*8 .. +7
            A.q[1] = *(const f32x4*)(xrow + 16 + h * 8);   // K = ks+16+h*8 .. +7
#pragma unroll
            for (int nt = 0; nt < 2; ++nt) {
                // B fragment (32x16 bf16): lane lm = column, two 8-elem K runs (matching A)
                const __bf16* brow =
                    &WT[buf][(nt * 16 + lm) * KP + ks + h * 8];
                union { v16bf v; f32x4 q[2]; } Bv;
                Bv.q[0] = *(const f32x4*)(brow);
                Bv.q[1] = *(const f32x4*)(brow + 16);
                if (nt == 0) {
                    acc0 = __builtin_amdgcn_wmma_f32_16x16x32_bf16(
                        false, A.v, false, Bv.v, (short)0, acc0, false, false);
                } else {
                    acc1 = __builtin_amdgcn_wmma_f32_16x16x32_bf16(
                        false, A.v, false, Bv.v, (short)0, acc1, false, false);
                }
            }
        }
        __syncthreads();       // next chunk staged & current reads complete
        buf ^= 1;
    }

    // ---- epilogue: + bias, permute to out[b, v, t] ----
#pragma unroll
    for (int nt = 0; nt < 2; ++nt) {
        int n = n0 + nt * 16 + lm;
        if (n < NCOL) {
            int t = n / NTOK;
            int v = n - t * NTOK;
            float bs = bias[n];
            v8f acc = nt ? acc1 : acc0;
#pragma unroll
            for (int r = 0; r < 8; ++r) {
                int b = wave * 16 + h * 8 + r;   // C layout: lanes16-31 hold M+8
                out[(size_t)b * NCOL + (size_t)v * L1 + t] = acc[r] + bs;
            }
        }
    }
}

extern "C" void kernel_launch(void* const* d_in, const int* in_sizes, int n_in,
                              void* d_out, int out_size, void* d_ws, size_t ws_size,
                              hipStream_t stream) {
    (void)in_sizes; (void)n_in; (void)out_size; (void)ws_size;
    const int*   src  = (const int*)d_in[0];     // (B, L1) token ids
    const float* emb  = (const float*)d_in[1];   // (6, 64)
    const float* W    = (const float*)d_in[2];   // (513, 64, 513, 6)
    const float* bias = (const float*)d_in[3];   // (513, 6)
    float* out = (float*)d_out;                  // (B, 6, 513)
    __bf16* X = (__bf16*)d_ws;                   // bf16 X[128][32832], 8.4 MB

    gather_x_kernel<<<(BATCH * L1 + 255) / 256, 256, 0, stream>>>(src, emb, X);
    gemm_wmma_kernel<<<(NCOL + NT - 1) / NT, 256, 0, stream>>>(X, W, bias, out);
}
